// HMultiheadAttention_49538152792297
// MI455X (gfx1250) — matrix operand order repaired and checked by reading, not compiled, not run on hardware
//
#include <hip/hip_runtime.h>
#include <math.h>

// ---------------- problem constants ----------------
#define Bc   4
#define Sc   1024
#define Ec   1024
#define Hc   16
#define Dc   64
#define NTOK (Bc * Sc)          // 4096 token rows
#define Cc   1.0f               // curvature
#define CSc  1.0f               // sqrt(C)
#define EPSc 1e-7f
// beta(E/2,1/2)/beta(D/2,1/2) = [G(32.5)/G(32)]/[G(512.5)/G(512)] = 4.0146775...
#define SCALE_BETA 4.0146775f

typedef __attribute__((ext_vector_type(16))) __bf16        v16bf;
typedef __attribute__((ext_vector_type(8)))  float         v8f;
typedef __attribute__((ext_vector_type(8)))  unsigned short us8;

// ---------------- small helpers ----------------
static __device__ __forceinline__ unsigned short f2bf(float f) {
  unsigned int u = __builtin_bit_cast(unsigned int, f);
  u += 0x7FFFu + ((u >> 16) & 1u);          // round-to-nearest-even
  return (unsigned short)(u >> 16);
}
// build a v16bf fragment from two contiguous 16-byte runs
union vfrag { v16bf v; us8 h[2]; };
static __device__ __forceinline__ v16bf ldfrag(const unsigned short* p0,
                                               const unsigned short* p1) {
  vfrag f;
  f.h[0] = *(const us8*)p0;   // -> ds_load_b128 / global_load_b128
  f.h[1] = *(const us8*)p1;
  return f.v;
}
// sum / max across the 16-lane half that shares an M-group (wave32)
static __device__ __forceinline__ float red16_sum(float v) {
  v += __shfl_xor(v, 1, 32); v += __shfl_xor(v, 2, 32);
  v += __shfl_xor(v, 4, 32); v += __shfl_xor(v, 8, 32);
  return v;
}
static __device__ __forceinline__ float red16_max(float v) {
  v = fmaxf(v, __shfl_xor(v, 1, 32)); v = fmaxf(v, __shfl_xor(v, 2, 32));
  v = fmaxf(v, __shfl_xor(v, 4, 32)); v = fmaxf(v, __shfl_xor(v, 8, 32));
  return v;
}
static __device__ __forceinline__ v8f wmma_bf16(v16bf a, v16bf b, v8f c) {
  return __builtin_amdgcn_wmma_f32_16x16x32_bf16(false, a, false, b, (short)0, c,
                                                 false, false);
}
// CDNA5 async global->LDS 16B copy (GVS addressing), tracked by ASYNCcnt
static __device__ __forceinline__ void async_copy16(unsigned lds_off,
                                                    const void* gbase,
                                                    unsigned goff) {
  asm volatile("global_load_async_to_lds_b128 %0, %1, %2"
               :
               : "v"(lds_off), "v"(goff), "s"((unsigned long long)gbase)
               : "memory");
}
static __device__ __forceinline__ void wait_async() {
  asm volatile("s_wait_asynccnt 0x0" ::: "memory");
}

// ---------------- stage 0a: token rows -> lambda + bf16 ----------------
__global__ __launch_bounds__(256) void k_row_prep(const float* __restrict__ X,
                                                  unsigned short* __restrict__ Xbf,
                                                  float* __restrict__ lamX) {
  int row = blockIdx.x;
  const float* xr = X + (size_t)row * Ec;
  unsigned short* xb = Xbf + (size_t)row * Ec;
  float s = 0.f;
  for (int c = threadIdx.x; c < Ec; c += 256) {
    float v = xr[c];
    s += v * v;
    xb[c] = f2bf(v);
  }
  __shared__ float red[8];
  for (int m = 1; m <= 16; m <<= 1) s += __shfl_xor(s, m, 32);
  if ((threadIdx.x & 31) == 0) red[threadIdx.x >> 5] = s;
  __syncthreads();
  if (threadIdx.x == 0) {
    float t = 0.f;
    for (int i = 0; i < 8; i++) t += red[i];
    lamX[row] = 2.f / fmaxf(1.f - Cc * t, EPSc);
  }
}

// ------- stage 0b: weight column norms + bf16 TRANSPOSED (K innermost) -------
__global__ __launch_bounds__(256) void k_col_prep(const float* __restrict__ Z,
                                                  unsigned short* __restrict__ ZbfT,
                                                  float* __restrict__ zn) {
  int col = blockIdx.x;                     // output unit
  float s = 0.f;
  for (int r = threadIdx.x; r < Ec; r += 256) {
    float v = Z[(size_t)r * Ec + col];
    s += v * v;
    ZbfT[(size_t)col * Ec + r] = f2bf(v);   // ZbfT[out][in]
  }
  __shared__ float red[8];
  for (int m = 1; m <= 16; m <<= 1) s += __shfl_xor(s, m, 32);
  if ((threadIdx.x & 31) == 0) red[threadIdx.x >> 5] = s;
  __syncthreads();
  if (threadIdx.x == 0) {
    float t = 0.f;
    for (int i = 0; i < 8; i++) t += red[i];
    zn[col] = fmaxf(sqrtf(t), 1e-15f);
  }
}

__global__ __launch_bounds__(256) void k_zero(float* __restrict__ p, int n) {
  int i = blockIdx.x * 256 + threadIdx.x;
  if (i < n) p[i] = 0.f;
}

// ------ stage 1: async double-buffered WMMA GEMM + hlinear epilogue ------
__global__ __launch_bounds__(256) void k_proj(const unsigned short* __restrict__ Xbf,
                                              const unsigned short* __restrict__ ZbfT,
                                              const float* __restrict__ lamX,
                                              const float* __restrict__ zn,
                                              const float* __restrict__ bias,
                                              float* __restrict__ W,
                                              float* __restrict__ wsq) {
  __shared__ __align__(16) unsigned short As[2][128][32];   // [row][K]  16 KB
  __shared__ __align__(16) unsigned short Bst[2][128][32];  // [col][K]  16 KB
  int tid = threadIdx.x, lane = tid & 31, wv = tid >> 5;
  int rowBase = blockIdx.x * 128, colBase = blockIdx.y * 128;
  int hk = lane >> 4;                        // lane half

  // each tile is 128 rows x 32 ushort = 512 x 16B; 256 threads -> 2 each
  auto issue_ab = [&](int b2, int kkn) {
#pragma unroll
    for (int t = 0; t < 2; t++) {
      int u = tid + t * 256;
      int r = u >> 2, c = u & 3;
      unsigned aoff = (unsigned)((((size_t)(rowBase + r)) * Ec + kkn) * 2u) +
                      (unsigned)c * 16u;
      async_copy16((unsigned)(uintptr_t)&As[b2][r][0] + (unsigned)c * 16u,
                   Xbf, aoff);
      unsigned boff = (unsigned)((((size_t)(colBase + r)) * Ec + kkn) * 2u) +
                      (unsigned)c * 16u;
      async_copy16((unsigned)(uintptr_t)&Bst[b2][r][0] + (unsigned)c * 16u,
                   ZbfT, boff);
    }
  };

  v8f acc[8];
#pragma unroll
  for (int n = 0; n < 8; n++) acc[n] = (v8f)0.f;

  issue_ab(0, 0);
  wait_async();
  __syncthreads();
  int bi = 0;

  int am = wv * 16 + (lane & 15);
  for (int kk = 0; kk < Ec; kk += 32) {
    if (kk + 32 < Ec) issue_ab(bi ^ 1, kk + 32);

    // preload A frag + all 8 B frags, then issue the 8 WMMAs back-to-back
    v16bf a = ldfrag(&As[bi][am][hk * 8], &As[bi][am][16 + hk * 8]);
    v16bf bfr[8];
#pragma unroll
    for (int n = 0; n < 8; n++) {
      int bn = n * 16 + (lane & 15);
      bfr[n] = ldfrag(&Bst[bi][bn][hk * 8], &Bst[bi][bn][16 + hk * 8]);
    }
#pragma unroll
    for (int n = 0; n < 8; n++) acc[n] = wmma_bf16(a, bfr[n], acc[n]);

    wait_async();      // next-tile async copies have landed
    __syncthreads();   // everyone finished reading buffer bi
    bi ^= 1;
  }

  // hlinear epilogue
  int rg = hk << 3;
  float psum[8], lamr[8];
#pragma unroll
  for (int r = 0; r < 8; r++) {
    psum[r] = 0.f;
    lamr[r] = lamX[rowBase + wv * 16 + r + rg];
  }
#pragma unroll
  for (int n = 0; n < 8; n++) {
    int col = colBase + n * 16 + (lane & 15);
    float znc = zn[col];
    float tcr = 2.f * CSc * bias[col];
    float ch = coshf(tcr), sh = sinhf(tcr);
#pragma unroll
    for (int r = 0; r < 8; r++) {
      float lam = lamr[r];
      float xz = acc[n][r];
      float vv = (2.f * znc / CSc) *
                 asinhf(CSc * lam * xz / znc * ch - (lam - 1.f) * sh);
      float wval = sinhf(CSc * vv) / CSc;
      int row = rowBase + wv * 16 + r + rg;
      W[(size_t)row * Ec + col] = wval;
      psum[r] += wval * wval;
    }
  }
#pragma unroll
  for (int r = 0; r < 8; r++) {
    float s = red16_sum(psum[r]);
    if ((lane & 15) == 0)
      atomicAdd(&wsq[rowBase + wv * 16 + r + rg], s);
  }
}

// ------- stage 2: normalize + repack (+V transposed [bh][d][s]) + stats -------
__global__ __launch_bounds__(256) void k_postproj(const float* __restrict__ W,
                                                  const float* __restrict__ wsq,
                                                  unsigned short* __restrict__ Pbf,
                                                  float* __restrict__ stat,
                                                  int vmode) {  // 0: q/k, 1: value
  int row = blockIdx.x;                     // b*S + s
  int b = row >> 10, s = row & (Sc - 1);
  float inv = 1.f / (1.f + sqrtf(1.f + Cc * wsq[row]));
  __shared__ float hs[Hc];
  if (threadIdx.x < Hc) hs[threadIdx.x] = 0.f;
  __syncthreads();
  for (int c = threadIdx.x; c < Ec; c += 256) {
    float p = W[(size_t)row * Ec + c] * inv;
    int h = c >> 6, d = c & 63;
    size_t idx = vmode
        ? (((size_t)(b * Hc + h)) * Dc + d) * Sc + s      // Vt[bh][d][s]
        : (((size_t)(b * Hc + h)) * Sc + s) * Dc + d;     // Q/K [bh][s][d]
    Pbf[idx] = f2bf(p);
    atomicAdd(&hs[h], p * p);
  }
  __syncthreads();
  if (threadIdx.x < Hc) {
    float sq = hs[threadIdx.x];
    float v = vmode ? (2.f / fmaxf(1.f - Cc * sq, EPSc)) : sq;
    stat[((size_t)(b * Hc + threadIdx.x)) * Sc + s] = v;
  }
}

// ---------------- stage 3: hyperbolic flash attention ----------------
// softmax normalizer cancels in num/dnm of the gyromidpoint -> only track row max
__global__ __launch_bounds__(256) void k_attn(const unsigned short* __restrict__ Qbf,
                                              const unsigned short* __restrict__ Kbf,
                                              const unsigned short* __restrict__ Vt,
                                              const float* __restrict__ q2g,
                                              const float* __restrict__ k2g,
                                              const float* __restrict__ lvg,
                                              float* __restrict__ tv) {
  __shared__ __align__(16) unsigned short Ks[2][32][Dc];   // [key][d]     8 KB
  __shared__ __align__(16) unsigned short Vst[2][Dc][32];  // [d][key]     8 KB
  __shared__ float k2sh[2][32], lvsh[2][32];
  __shared__ __align__(16) unsigned short pls[8][16][32];  // [qrow][key]  8 KB

  int bh = blockIdx.y;
  int qbase = blockIdx.x * 128;
  int tid = threadIdx.x, lane = tid & 31, wv = tid >> 5;
  int hk = lane >> 4, rg = hk << 3;

  // double-buffered async tile copy (K: 4KB contiguous, V: 64 rows x 64B)
  auto issue_tile = [&](int b2, int ktn) {
    unsigned kof = (unsigned)(((size_t)bh * Sc + ktn) * Dc * 2u);
    async_copy16((unsigned)(uintptr_t)&Ks[b2][0][0] + (unsigned)tid * 16u,
                 Kbf, kof + (unsigned)tid * 16u);
    int d = tid >> 2, ch = tid & 3;
    unsigned vof = (unsigned)((((size_t)bh * Dc + d) * Sc + ktn) * 2u) +
                   (unsigned)ch * 16u;
    async_copy16((unsigned)(uintptr_t)&Vst[b2][0][0] +
                     (unsigned)(d * 64 + ch * 16),
                 Vt, vof);
    if (tid < 32) {
      k2sh[b2][tid] = k2g[(size_t)bh * Sc + ktn + tid];
      lvsh[b2][tid] = lvg[(size_t)bh * Sc + ktn + tid];
    }
  };

  // Q fragments for this wave's 16 rows (depth chunks K=0..31, 32..63)
  const unsigned short* Qp =
      Qbf + ((size_t)bh * Sc + qbase + wv * 16 + (lane & 15)) * Dc;
  v16bf qa[2];
#pragma unroll
  for (int kc = 0; kc < 2; kc++)
    qa[kc] = ldfrag(Qp + kc * 32 + hk * 8, Qp + kc * 32 + 16 + hk * 8);

  float q2r[8];
#pragma unroll
  for (int r = 0; r < 8; r++)
    q2r[r] = q2g[(size_t)bh * Sc + qbase + wv * 16 + r + rg];

  float mrow[8], dnm[8];
  v8f num[4];
#pragma unroll
  for (int r = 0; r < 8; r++) { mrow[r] = -1e30f; dnm[r] = 0.f; }
#pragma unroll
  for (int dt = 0; dt < 4; dt++) num[dt] = (v8f)0.f;

  issue_tile(0, 0);
  wait_async();
  __syncthreads();
  int bi = 0;

  for (int kt = 0; kt < Sc; kt += 32) {
    if (kt + 32 < Sc) issue_tile(bi ^ 1, kt + 32);

    // QK^T: 16q x 32k — preload 4 B frags, then 4 WMMAs back-to-back
    v16bf bkf[4];
#pragma unroll
    for (int ks = 0; ks < 2; ks++) {
      int bn = ks * 16 + (lane & 15);
#pragma unroll
      for (int dc = 0; dc < 2; dc++)
        bkf[ks * 2 + dc] = ldfrag(&Ks[bi][bn][dc * 32 + hk * 8],
                                  &Ks[bi][bn][dc * 32 + 16 + hk * 8]);
    }
    v8f sacc[2];
#pragma unroll
    for (int ks = 0; ks < 2; ks++) sacc[ks] = (v8f)0.f;
#pragma unroll
    for (int ks = 0; ks < 2; ks++)
#pragma unroll
      for (int dc = 0; dc < 2; dc++)
        sacc[ks] = wmma_bf16(qa[dc], bkf[ks * 2 + dc], sacc[ks]);

    // scores -> Poincare distance -> online max rescale
    float p0[8], p1[8];
#pragma unroll
    for (int r = 0; r < 8; r++) {
      float q2v = q2r[r];
      float oq = 1.f - Cc * q2v;
#pragma unroll
      for (int ks = 0; ks < 2; ks++) {
        float qk = (ks == 0 ? sacc[0][r] : sacc[1][r]);
        float k2v = k2sh[bi][ks * 16 + (lane & 15)];
        float diff2 = fmaxf(q2v + k2v - 2.f * qk, 0.f);
        float den = fmaxf(oq * (1.f - Cc * k2v), EPSc);
        float arg = fmaxf(1.f + 2.f * Cc * diff2 / den, 1.f + 1e-6f);
        float sc = -acoshf(arg) / CSc;
        if (ks == 0) p0[r] = sc; else p1[r] = sc;
      }
    }
#pragma unroll
    for (int r = 0; r < 8; r++) {
      float lm = red16_max(fmaxf(p0[r], p1[r]));
      float mnew = fmaxf(mrow[r], lm);
      float scale = __expf(mrow[r] - mnew);
      mrow[r] = mnew;
      dnm[r] *= scale;
#pragma unroll
      for (int dt = 0; dt < 4; dt++) num[dt][r] *= scale;
      float e0 = __expf(p0[r] - mnew);
      float e1 = __expf(p1[r] - mnew);
      int k0 = lane & 15, k1 = 16 + (lane & 15);
      float lv0 = lvsh[bi][k0], lv1 = lvsh[bi][k1];
      dnm[r] += red16_sum(e0 * (lv0 - 1.f) + e1 * (lv1 - 1.f));
      pls[wv][r + rg][k0] = f2bf(e0 * lv0);
      pls[wv][r + rg][k1] = f2bf(e1 * lv1);
    }
    __syncthreads();   // per-wave score tiles visible cross-lane

    // PV: (16q x 32k) @ (32k x 64d) — preload 4 B frags, 4 WMMAs back-to-back
    v16bf apl = ldfrag(&pls[wv][lane & 15][hk * 8],
                       &pls[wv][lane & 15][16 + hk * 8]);
    v16bf bvf[4];
#pragma unroll
    for (int dt = 0; dt < 4; dt++) {
      int dn = dt * 16 + (lane & 15);
      bvf[dt] = ldfrag(&Vst[bi][dn][hk * 8], &Vst[bi][dn][16 + hk * 8]);
    }
#pragma unroll
    for (int dt = 0; dt < 4; dt++) num[dt] = wmma_bf16(apl, bvf[dt], num[dt]);

    wait_async();      // this wave's next-tile async copies have landed
    __syncthreads();   // everyone finished reading buffer bi
    bi ^= 1;
  }

  // finalize: gyromidpoint -> logmap0 -> beta scale, write tangent [B,S,E]
  int b = bh >> 4, h = bh & (Hc - 1);
#pragma unroll
  for (int r = 0; r < 8; r++) {
    float dn = fmaxf(dnm[r], EPSc);
    float uu[4], u2 = 0.f;
#pragma unroll
    for (int dt = 0; dt < 4; dt++) {
      uu[dt] = num[dt][r] / dn;
      u2 += uu[dt] * uu[dt];
    }
    u2 = red16_sum(u2);
    float inv = 1.f / (1.f + sqrtf(fmaxf(1.f - Cc * u2, EPSc)));
    float nm = sqrtf(fmaxf(u2 * inv * inv, 1e-15f));  // |mid|
    float t = fminf(CSc * nm, 1.f - 1e-6f);
    float fac = atanhf(t) / (CSc * nm) * inv * SCALE_BETA;
    int srow = qbase + wv * 16 + r + rg;
#pragma unroll
    for (int dt = 0; dt < 4; dt++) {
      int d = dt * 16 + (lane & 15);
      tv[((size_t)(b * Sc + srow)) * Ec + h * Dc + d] = uu[dt] * fac;
    }
  }
}

// ---------------- stage 4: expmap0 over full embedding ----------------
__global__ __launch_bounds__(256) void k_expmap(const float* __restrict__ tvb,
                                                float* __restrict__ out) {
  int row = blockIdx.x;
  __shared__ float red[8];
  float s = 0.f;
  for (int c = threadIdx.x; c < Ec; c += 256) {
    float v = tvb[(size_t)row * Ec + c];
    s += v * v;
  }
  for (int m = 1; m <= 16; m <<= 1) s += __shfl_xor(s, m, 32);
  if ((threadIdx.x & 31) == 0) red[threadIdx.x >> 5] = s;
  __syncthreads();
  float tot = 0.f;
  for (int i = 0; i < 8; i++) tot += red[i];
  float n = sqrtf(fmaxf(tot, 1e-15f));
  float fac = tanhf(CSc * n) / (CSc * n);
  for (int c = threadIdx.x; c < Ec; c += 256)
    out[(size_t)row * Ec + c] = fac * tvb[(size_t)row * Ec + c];
}

// ---------------- host orchestration ----------------
extern "C" void kernel_launch(void* const* d_in, const int* in_sizes, int n_in,
                              void* d_out, int out_size, void* d_ws, size_t ws_size,
                              hipStream_t stream) {
  const float* Xin[3] = { (const float*)d_in[0], (const float*)d_in[1],
                          (const float*)d_in[2] };                     // q,k,v
  const float* Zin[3] = { (const float*)d_in[3], (const float*)d_in[5],
                          (const float*)d_in[7] };                     // zq,zk,zv
  const float* Bin[3] = { (const float*)d_in[4], (const float*)d_in[6],
                          (const float*)d_in[8] };                     // bq,bk,bv

  // workspace layout (bytes)
  char* ws = (char*)d_ws;
  size_t off = 0;
  unsigned short* Xbf  = (unsigned short*)(ws + off); off += (size_t)3 * NTOK * Ec * 2;
  unsigned short* ZbfT = (unsigned short*)(ws + off); off += (size_t)3 * Ec * Ec * 2;
  float* lamX = (float*)(ws + off); off += (size_t)3 * NTOK * 4;
  float* znA  = (float*)(ws + off); off += (size_t)3 * Ec * 4;
  float* Wbuf = (float*)(ws + off); off += (size_t)3 * NTOK * Ec * 4;
  float* wsq  = (float*)(ws + off); off += (size_t)3 * NTOK * 4;
  unsigned short* Pbf = (unsigned short*)(ws + off); off += (size_t)3 * NTOK * Ec * 2;
  float* stat = (float*)(ws + off); off += (size_t)3 * Bc * Hc * Sc * 4;
  float* tvb  = (float*)(ws + off); off += (size_t)NTOK * Ec * 4;
  (void)ws_size; (void)n_in; (void)in_sizes; (void)out_size;

  k_zero<<<(3 * NTOK + 255) / 256, 256, 0, stream>>>(wsq, 3 * NTOK);

  for (int s = 0; s < 3; s++) {
    k_row_prep<<<NTOK, 256, 0, stream>>>(Xin[s], Xbf + (size_t)s * NTOK * Ec,
                                         lamX + (size_t)s * NTOK);
    k_col_prep<<<Ec, 256, 0, stream>>>(Zin[s], ZbfT + (size_t)s * Ec * Ec,
                                       znA + (size_t)s * Ec);
  }
  for (int s = 0; s < 3; s++) {
    dim3 g(NTOK / 128, Ec / 128);
    k_proj<<<g, 256, 0, stream>>>(Xbf + (size_t)s * NTOK * Ec,
                                  ZbfT + (size_t)s * Ec * Ec,
                                  lamX + (size_t)s * NTOK, znA + (size_t)s * Ec,
                                  Bin[s], Wbuf + (size_t)s * NTOK * Ec,
                                  wsq + (size_t)s * NTOK);
  }
  for (int s = 0; s < 3; s++) {
    k_postproj<<<NTOK, 256, 0, stream>>>(Wbuf + (size_t)s * NTOK * Ec,
                                         wsq + (size_t)s * NTOK,
                                         Pbf + (size_t)s * NTOK * Ec,
                                         stat + (size_t)s * Bc * Hc * Sc,
                                         s == 2 ? 1 : 0);
  }
  {
    dim3 g(Sc / 128, Bc * Hc);
    k_attn<<<g, 256, 0, stream>>>(Pbf, Pbf + (size_t)1 * NTOK * Ec,
                                  Pbf + (size_t)2 * NTOK * Ec,
                                  stat, stat + (size_t)1 * Bc * Hc * Sc,
                                  stat + (size_t)2 * Bc * Hc * Sc, tvb);
  }
  k_expmap<<<NTOK, 256, 0, stream>>>(tvb, (float*)d_out);
}
